// ModelWithAttention_11871289606792
// MI455X (gfx1250) — compile-verified, matrix operand order checked by reading, not run on hardware
//
#include <hip/hip_runtime.h>
#include <hip/hip_bf16.h>
#include <math.h>

// ---------------- types ----------------
typedef __bf16 bf16_t;
typedef __attribute__((ext_vector_type(16))) __bf16 bf16x16;
typedef __attribute__((ext_vector_type(8)))  __bf16 bf16x8;
typedef __attribute__((ext_vector_type(8)))  float  floatx8;

// ---------------- problem dims ----------------
#define NB 64       // batch
#define HH 512      // hidden
#define AA 256      // attention dim
#define VV 10000    // vocab
#define TT 151      // seq length
#define LL 49       // encoder positions (7*7)
#define LP 64       // LL padded to multiple of 16

// =====================================================================
// WMMA fragment helpers (bf16 16x16x32, wave32 layouts per CDNA5 ISA)
// A operand: lane m(0..15)=row, VGPR0..3 hold K k0+0..7 (pairs),
// VGPR4..7 hold K k0+16..23; lanes 16..31 same rows, K +8.
// B operand (SRC1): identical indexing by output column n over [N][K] data.
// =====================================================================
__device__ __forceinline__ bf16x16 ld_frag(const bf16_t* __restrict__ base,
                                           int ld, int row, int k0, int lane) {
  const bf16_t* p = base + (size_t)row * (size_t)ld + k0 + ((lane & 16) ? 8 : 0);
  bf16x8 lo = *(const bf16x8*)(p);
  bf16x8 hi = *(const bf16x8*)(p + 16);
  bf16x16 r;
#pragma unroll
  for (int i = 0; i < 8; ++i) { r[i] = lo[i]; r[i + 8] = hi[i]; }
  return r;
}

// =====================================================================
// Generic batched GEMM: C[M,N] = act(A[M,K] @ Bt[N,K]^T + bias)
//   A   : bf16 row-major [M][K], batch stride sA (elements)
//   Bt  : bf16 weights stored [N][K] ("column major")
//   C   : optional fp32 out (rows < crows stored); omode 1 => [B][V][T] scatter
//   Cb  : optional bf16 out (rows < cbrows stored)
// One wave (32 thr) per block computes a (16*MSUB) x (16*NSUB) tile:
// MSUB*NSUB independent accumulators, MSUB+NSUB fragment loads per K-chunk.
// MSUB/NSUB/KDIM are compile-time: no guards near the WMMAs, K-loop fully
// unrolled so loads pipeline ahead of the WMMAs.
// =====================================================================
template<int MSUB, int NSUB, int KDIM>
__global__ __launch_bounds__(32)
void wmma_gemm(const bf16_t* __restrict__ A, long long sA, int lda,
               const bf16_t* __restrict__ Bt, int ldb, int n_off,
               const float* __restrict__ bias,
               float* __restrict__ C, long long sC, int ldc, int crows,
               bf16_t* __restrict__ Cb, long long sCb, int ldcb, int cbrows,
               int act, int omode, int Tdim)
{
  const int lane   = threadIdx.x & 31;
  const int batch  = blockIdx.z;
  const int tileM  = blockIdx.y * (16 * MSUB);
  const int tileN0 = n_off + blockIdx.x * (16 * NSUB);
  const bf16_t* Ab = A + (size_t)batch * (size_t)sA;

  floatx8 acc[MSUB][NSUB] = {};

#pragma unroll
  for (int kk = 0; kk < KDIM / 32; ++kk) {
    const int k0 = kk * 32;
    bf16x16 af[MSUB];
#pragma unroll
    for (int mi = 0; mi < MSUB; ++mi)
      af[mi] = ld_frag(Ab, lda, tileM + 16 * mi + (lane & 15), k0, lane);
    bf16x16 bfr[NSUB];
#pragma unroll
    for (int i = 0; i < NSUB; ++i)
      bfr[i] = ld_frag(Bt, ldb, tileN0 + 16 * i + (lane & 15), k0, lane);
#pragma unroll
    for (int mi = 0; mi < MSUB; ++mi)
#pragma unroll
      for (int i = 0; i < NSUB; ++i)
        acc[mi][i] = __builtin_amdgcn_wmma_f32_16x16x32_bf16(
            false, af[mi], false, bfr[i], (short)0, acc[mi][i], false, false);
  }

#pragma unroll
  for (int mi = 0; mi < MSUB; ++mi) {
    const int rbase = tileM + 16 * mi + ((lane & 16) ? 8 : 0);
#pragma unroll
    for (int i = 0; i < NSUB; ++i) {
      int col = tileN0 + 16 * i + (lane & 15);
      float bv = bias ? bias[col] : 0.0f;
#pragma unroll
      for (int j = 0; j < 8; ++j) {
        int row = rbase + j;
        float v = acc[mi][i][j] + bv;
        if (act == 1) v = tanhf(v);
        if (C && row < crows) {
          if (omode == 0) {
            C[(size_t)batch * (size_t)sC + (size_t)row * (size_t)ldc + col] = v;
          } else {          // [B][V][T] scatter: row = b*T + t, col = v
            int bb = row / Tdim, tt = row % Tdim;
            C[((size_t)bb * (size_t)ldc + col) * (size_t)Tdim + tt] = v;
          }
        }
        if (Cb && row < cbrows) {
          Cb[(size_t)batch * (size_t)sCb + (size_t)row * (size_t)ldcb + col] = (bf16_t)v;
        }
      }
    }
  }
}

// ---------------- weight prep ----------------
__global__ void convT_kernel(const float* __restrict__ in, bf16_t* __restrict__ out,
                             int K, int N) { // in [K][N] -> out [N][K]
  int idx = blockIdx.x * blockDim.x + threadIdx.x;
  if (idx < K * N) {
    int n = idx / K, k = idx % K;
    out[idx] = (bf16_t)in[(size_t)k * N + n];
  }
}
__global__ void convC_kernel(const float* __restrict__ in, bf16_t* __restrict__ out, int n) {
  int idx = blockIdx.x * blockDim.x + threadIdx.x;
  if (idx < n) out[idx] = (bf16_t)in[idx];
}
// X[b][l][k] = enc_feat[b][k][l] (transpose), zero-padded rows l=49..63
__global__ void buildX_kernel(const float* __restrict__ enc_feat, bf16_t* __restrict__ X) {
  int idx = blockIdx.x * blockDim.x + threadIdx.x;
  if (idx >= NB * LP * HH) return;
  int b = idx / (LP * HH);
  int r = idx % (LP * HH);
  int l = r / HH, k = r % HH;
  X[idx] = (l < LL) ? (bf16_t)enc_feat[((size_t)b * HH + k) * LL + l] : (bf16_t)0.0f;
}

// ---------------- small fused kernels ----------------
__global__ void h0_kernel(const float* __restrict__ enc, float* __restrict__ h,
                          bf16_t* __restrict__ hB) {
  int b = blockIdx.x, j = threadIdx.x;
  float s = 0.0f;
  for (int l = 0; l < LL; ++l) s += enc[((size_t)b * LL + l) * HH + j];
  s *= (1.0f / 49.0f);
  h[(size_t)b * HH + j] = s;
  hB[(size_t)b * HH + j] = (bf16_t)s;
}

__global__ void seq0_kernel(const float* __restrict__ embed, const int* __restrict__ start,
                            bf16_t* __restrict__ seqB) {
  int b = blockIdx.x, j = threadIdx.x;
  float v = embed[(size_t)(*start) * HH + j];
  seqB[((size_t)b * TT) * HH + j] = (bf16_t)v;
}

// attention energies + softmax + context; also assembles [inp | context] bf16 row
__global__ void attn_kernel(const float* __restrict__ att1,  // [B][49][256]
                            const float* __restrict__ att2,  // [B][256]
                            const float* __restrict__ Wfull, // [256]
                            const float* __restrict__ bfull, // [1]
                            const float* __restrict__ enc,   // [B][49][512]
                            const bf16_t* __restrict__ seqB, // [B*T][512]
                            int t,
                            bf16_t* __restrict__ catB)       // [B][1024]
{
  int b = blockIdx.x;
  int tid = threadIdx.x;          // 0..255
  int lane = tid & 31;
  int wv = tid >> 5;              // 0..7
  __shared__ float sh_a[LL];

  const float* a2 = att2 + (size_t)b * AA;
  const float* a1 = att1 + (size_t)b * LL * AA;
  for (int l = wv; l < LL; l += 8) {
    float p = 0.0f;
#pragma unroll
    for (int a = lane; a < AA; a += 32)
      p += tanhf(a1[(size_t)l * AA + a] + a2[a]) * Wfull[a];
    for (int m = 16; m > 0; m >>= 1) p += __shfl_xor(p, m);
    if (lane == 0) sh_a[l] = p + bfull[0];
  }
  __syncthreads();

  if (wv == 0) {                  // softmax over 49 by wave 0
    float v0 = (lane < LL) ? sh_a[lane] : -3.4e38f;
    float v1 = (lane + 32 < LL) ? sh_a[lane + 32] : -3.4e38f;
    float mx = fmaxf(v0, v1);
    for (int m = 16; m > 0; m >>= 1) mx = fmaxf(mx, __shfl_xor(mx, m));
    float e0 = (lane < LL) ? __expf(v0 - mx) : 0.0f;
    float e1 = (lane + 32 < LL) ? __expf(v1 - mx) : 0.0f;
    float s = e0 + e1;
    for (int m = 16; m > 0; m >>= 1) s += __shfl_xor(s, m);
    float inv = 1.0f / s;
    if (lane < LL) sh_a[lane] = e0 * inv;
    if (lane + 32 < LL) sh_a[lane + 32] = e1 * inv;
  }
  __syncthreads();

  const float* eb = enc + (size_t)b * LL * HH;
#pragma unroll
  for (int r = 0; r < 2; ++r) {
    int hh = tid + r * 256;
    float c = 0.0f;
    for (int l = 0; l < LL; ++l) c += sh_a[l] * eb[(size_t)l * HH + hh];
    catB[(size_t)b * (2 * HH) + HH + hh] = (bf16_t)c;
    catB[(size_t)b * (2 * HH) + hh] = seqB[((size_t)b * TT + t) * HH + hh];
  }
}

__global__ void gru_kernel(const float* __restrict__ gi, const float* __restrict__ gh,
                           float* __restrict__ h, bf16_t* __restrict__ hB,
                           bf16_t* __restrict__ seqB, int t) {
  int b = blockIdx.x, j = threadIdx.x;
  const float* gib = gi + (size_t)b * 3 * HH;
  const float* ghb = gh + (size_t)b * 3 * HH;
  float r = 1.0f / (1.0f + __expf(-(gib[j] + ghb[j])));
  float z = 1.0f / (1.0f + __expf(-(gib[HH + j] + ghb[HH + j])));
  float n = tanhf(gib[2 * HH + j] + r * ghb[2 * HH + j]);
  float hn = (1.0f - z) * n + z * h[(size_t)b * HH + j];
  h[(size_t)b * HH + j] = hn;
  hB[(size_t)b * HH + j] = (bf16_t)hn;
  seqB[((size_t)b * TT + (t + 1)) * HH + j] = (bf16_t)hn;
}

__global__ void argmax_kernel(const float* __restrict__ res, float* __restrict__ tok) {
  int m = blockIdx.x;               // b*T + t
  int b = m / TT, t = m % TT;
  int tid = threadIdx.x;
  const float* base = res + (size_t)b * VV * TT + t;
  float best = -3.4e38f; int bi = 0;
  for (int v = tid; v < VV; v += 256) {
    float x = base[(size_t)v * TT];
    if (x > best) { best = x; bi = v; }
  }
  __shared__ float sv[256]; __shared__ int si[256];
  sv[tid] = best; si[tid] = bi;
  __syncthreads();
  for (int s = 128; s > 0; s >>= 1) {
    if (tid < s) {
      if (sv[tid + s] > sv[tid] || (sv[tid + s] == sv[tid] && si[tid + s] < si[tid])) {
        sv[tid] = sv[tid + s]; si[tid] = si[tid + s];
      }
    }
    __syncthreads();
  }
  if (tid == 0) tok[m] = (float)si[0];
}

// =====================================================================
// host launch
// =====================================================================
static inline void* wsp(void* base, size_t& off, size_t bytes) {
  void* p = (char*)base + off;
  off += (bytes + 255) & ~(size_t)255;
  return p;
}
static inline int cdiv(int a, int b) { return (a + b - 1) / b; }

extern "C" void kernel_launch(void* const* d_in, const int* in_sizes, int n_in,
                              void* d_out, int out_size, void* d_ws, size_t ws_size,
                              hipStream_t stream) {
  (void)in_sizes; (void)n_in; (void)out_size; (void)ws_size;
  const float* enc_feat = (const float*)d_in[0];
  const float* W_fp   = (const float*)d_in[1];
  const float* b_fp   = (const float*)d_in[2];
  const float* W_enc  = (const float*)d_in[3];
  const float* b_enc  = (const float*)d_in[4];
  const float* W_dec  = (const float*)d_in[5];
  const float* b_dec  = (const float*)d_in[6];
  const float* W_full = (const float*)d_in[7];
  const float* b_full = (const float*)d_in[8];
  const float* W_comb = (const float*)d_in[9];
  const float* b_comb = (const float*)d_in[10];
  const float* embed  = (const float*)d_in[11];
  const float* W_ih   = (const float*)d_in[12];
  const float* b_ih   = (const float*)d_in[13];
  const float* W_hh   = (const float*)d_in[14];
  const float* b_hh   = (const float*)d_in[15];
  const float* W_proj = (const float*)d_in[16];
  const float* b_proj = (const float*)d_in[17];
  const int*   startp = (const int*)d_in[18];

  float* res = (float*)d_out;                              // [B][V][T]
  float* tok = (float*)d_out + (size_t)NB * VV * TT;       // [B][T]

  size_t off = 0;
  bf16_t* Wt_fp   = (bf16_t*)wsp(d_ws, off, (size_t)HH * HH * 2);     // [512][512]
  bf16_t* Wt_enc  = (bf16_t*)wsp(d_ws, off, (size_t)AA * HH * 2);     // [256][512]
  bf16_t* Wt_dec  = (bf16_t*)wsp(d_ws, off, (size_t)AA * HH * 2);     // [256][512]
  bf16_t* Wt_comb = (bf16_t*)wsp(d_ws, off, (size_t)HH * 2 * HH * 2); // [512][1024]
  bf16_t* W_ihb   = (bf16_t*)wsp(d_ws, off, (size_t)3 * HH * HH * 2); // [1536][512]
  bf16_t* W_hhb   = (bf16_t*)wsp(d_ws, off, (size_t)3 * HH * HH * 2);
  bf16_t* Wt_proj = (bf16_t*)wsp(d_ws, off, (size_t)VV * HH * 2);     // [10000][512]
  bf16_t* Xbf     = (bf16_t*)wsp(d_ws, off, (size_t)NB * LP * HH * 2);
  float*  encf    = (float*) wsp(d_ws, off, (size_t)NB * LL * HH * 4);
  bf16_t* encb    = (bf16_t*)wsp(d_ws, off, (size_t)NB * LP * HH * 2);
  float*  att1    = (float*) wsp(d_ws, off, (size_t)NB * LL * AA * 4);
  float*  att2    = (float*) wsp(d_ws, off, (size_t)NB * AA * 4);
  float*  hF      = (float*) wsp(d_ws, off, (size_t)NB * HH * 4);
  bf16_t* hB      = (bf16_t*)wsp(d_ws, off, (size_t)NB * HH * 2);
  bf16_t* catB    = (bf16_t*)wsp(d_ws, off, (size_t)NB * 2 * HH * 2);
  bf16_t* combB   = (bf16_t*)wsp(d_ws, off, (size_t)NB * HH * 2);
  float*  gi      = (float*) wsp(d_ws, off, (size_t)NB * 3 * HH * 4);
  float*  gh      = (float*) wsp(d_ws, off, (size_t)NB * 3 * HH * 4);
  bf16_t* seqB    = (bf16_t*)wsp(d_ws, off, (size_t)NB * TT * HH * 2);

  // ---- weight conversion (bf16, B-operand [N][K] layout) ----
  convT_kernel<<<cdiv(HH * HH, 256), 256, 0, stream>>>(W_fp,   Wt_fp,   HH,     HH);
  convT_kernel<<<cdiv(HH * AA, 256), 256, 0, stream>>>(W_enc,  Wt_enc,  HH,     AA);
  convT_kernel<<<cdiv(HH * AA, 256), 256, 0, stream>>>(W_dec,  Wt_dec,  HH,     AA);
  convT_kernel<<<cdiv(2 * HH * HH, 256), 256, 0, stream>>>(W_comb, Wt_comb, 2 * HH, HH);
  convC_kernel<<<cdiv(3 * HH * HH, 256), 256, 0, stream>>>(W_ih, W_ihb, 3 * HH * HH);
  convC_kernel<<<cdiv(3 * HH * HH, 256), 256, 0, stream>>>(W_hh, W_hhb, 3 * HH * HH);
  convT_kernel<<<cdiv(HH * VV, 256), 256, 0, stream>>>(W_proj, Wt_proj, HH,     VV);
  buildX_kernel<<<cdiv(NB * LP * HH, 256), 256, 0, stream>>>(enc_feat, Xbf);

  // ---- encoder projection: enc = X @ W_fp + b_fp  (batched) ----
  {
    dim3 g(HH / 64, LP / 16, NB);
    wmma_gemm<1, 4, HH><<<g, 32, 0, stream>>>(Xbf, (long long)LP * HH, HH, Wt_fp, HH, 0, b_fp,
                                              encf, (long long)LL * HH, HH, LL,
                                              encb, (long long)LP * HH, HH, LP,
                                              0, 0, 0);
  }
  h0_kernel<<<NB, HH, 0, stream>>>(encf, hF, hB);
  // ---- att1 = enc @ W_enc + b_enc (batched) ----
  {
    dim3 g(AA / 64, LP / 16, NB);
    wmma_gemm<1, 4, HH><<<g, 32, 0, stream>>>(encb, (long long)LP * HH, HH, Wt_enc, HH, 0, b_enc,
                                              att1, (long long)LL * AA, AA, LL,
                                              nullptr, 0, 0, 0,
                                              0, 0, 0);
  }
  seq0_kernel<<<NB, HH, 0, stream>>>(embed, startp, seqB);

  // ---- sequential decode: 150 steps ----
  for (int t = 0; t < TT - 1; ++t) {
    { // att2 = h @ W_dec + b_dec
      dim3 g(AA / 64, NB / 16, 1);
      wmma_gemm<1, 4, HH><<<g, 32, 0, stream>>>(hB, 0, HH, Wt_dec, HH, 0, b_dec,
                                                att2, 0, AA, NB, nullptr, 0, 0, 0,
                                                0, 0, 0);
    }
    attn_kernel<<<NB, 256, 0, stream>>>(att1, att2, W_full, b_full, encf, seqB, t, catB);
    { // comb = tanh([inp|ctx] @ W_comb + b_comb) -> bf16
      dim3 g(HH / 64, NB / 16, 1);
      wmma_gemm<1, 4, 2 * HH><<<g, 32, 0, stream>>>(catB, 0, 2 * HH, Wt_comb, 2 * HH, 0, b_comb,
                                                    nullptr, 0, 0, 0,
                                                    combB, 0, HH, NB,
                                                    1, 0, 0);
    }
    { // gi = comb @ W_ih^T + b_ih ; gh = h @ W_hh^T + b_hh
      dim3 g(3 * HH / 64, NB / 16, 1);
      wmma_gemm<1, 4, HH><<<g, 32, 0, stream>>>(combB, 0, HH, W_ihb, HH, 0, b_ih,
                                                gi, 0, 3 * HH, NB, nullptr, 0, 0, 0,
                                                0, 0, 0);
      wmma_gemm<1, 4, HH><<<g, 32, 0, stream>>>(hB, 0, HH, W_hhb, HH, 0, b_hh,
                                                gh, 0, 3 * HH, NB, nullptr, 0, 0, 0,
                                                0, 0, 0);
    }
    gru_kernel<<<NB, HH, 0, stream>>>(gi, gh, hF, hB, seqB, t);
  }

  // ---- final projection: res[b][v][t] = seq[b][t] @ W_proj + b_proj ----
  {
    // main body: 156 * 64 = 9984 columns, 64-row M tiles (9664 = 151 * 64)
    dim3 g(156, (NB * TT) / 64, 1);
    wmma_gemm<4, 4, HH><<<g, 32, 0, stream>>>(seqB, 0, HH, Wt_proj, HH, 0, b_proj,
                                              res, 0, VV, NB * TT, nullptr, 0, 0, 0,
                                              0, 1, TT);
    // tail: columns 9984..9999 (one 16-wide tile)
    dim3 gt(1, (NB * TT) / 64, 1);
    wmma_gemm<4, 1, HH><<<gt, 32, 0, stream>>>(seqB, 0, HH, Wt_proj, HH, 9984, b_proj,
                                               res, 0, VV, NB * TT, nullptr, 0, 0, 0,
                                               0, 1, TT);
  }
  argmax_kernel<<<NB * TT, 256, 0, stream>>>(res, tok);
}
